// RelativeAttention_65326452572233
// MI455X (gfx1250) — matrix operand order repaired and checked by reading, not compiled, not run on hardware
//
#include <hip/hip_runtime.h>
#include <hip/hip_bf16.h>

typedef __attribute__((ext_vector_type(16))) __bf16 v16bf;
typedef __attribute__((ext_vector_type(8)))  float  v8f;
typedef __attribute__((ext_vector_type(4)))  __bf16 v4bf;

#define B_  8
#define S_  4096
#define D_  512
#define H_  8
#define DH_ 64
#define W_  8
#define M_  (B_ * S_)        // 32768 rows of x
#define NT_ (3 * D_)         // 1536 fused output columns (q|k|v)

#define TILE_M 128
#define TILE_N 128
#define KB     32
#define KSTEPS (D_ / KB)     // 16
#define LDSK   40            // padded row stride (bf16 elems) -> 80B, 16B-aligned chunks
#define STAGE_ELEMS ((TILE_M + TILE_N) * LDSK)   // 10240 bf16 per stage

// ---------------------------------------------------------------- converts ---

__global__ __launch_bounds__(256) void convert_x_bf16(const float* __restrict__ x,
                                                      __bf16* __restrict__ xbf) {
  int i = blockIdx.x * blockDim.x + threadIdx.x;       // over M_*D_/4 float4s
  const float4 f = ((const float4*)x)[i];
  v4bf o;
  o.x = (__bf16)f.x; o.y = (__bf16)f.y; o.z = (__bf16)f.z; o.w = (__bf16)f.w;
  ((v4bf*)xbf)[i] = o;
}

// Build W^T (bf16), fused [1536][512]: row ng = which*512+n holds column n of W_which.
__global__ __launch_bounds__(256) void convert_wt_bf16(const float* __restrict__ Wq,
                                                       const float* __restrict__ Wk,
                                                       const float* __restrict__ Wv,
                                                       __bf16* __restrict__ wt) {
  int idx = blockIdx.x * blockDim.x + threadIdx.x;     // over NT_*D_
  int kk = idx & (D_ - 1);
  int ng = idx >> 9;
  int which = ng >> 9;
  int n = ng & (D_ - 1);
  const float* Wm = (which == 0) ? Wq : ((which == 1) ? Wk : Wv);
  wt[(size_t)ng * D_ + kk] = (__bf16)Wm[(size_t)kk * D_ + n];
}

// ------------------------------------------------------- CDNA5 async copy ----

// GLOBAL_LOAD_ASYNC_TO_LDS_B128: per-lane 16B, global -> LDS, tracked by ASYNCcnt.
__device__ __forceinline__ void async_copy16(unsigned ldsOff, const __bf16* gptr) {
  asm volatile("global_load_async_to_lds_b128 %0, %1, off"
               :: "v"(ldsOff), "v"(gptr)
               : "memory");
}

__device__ __forceinline__ void wait_async0() {
  asm volatile("s_wait_asynccnt 0x0" ::: "memory");
}

// ----------------------------------------------------------------- GEMM -----

__device__ __forceinline__ v16bf load_frag(const __bf16* p, int gapElems) {
  union { v16bf v; uint4 u[2]; } f;
  f.u[0] = *(const uint4*)(p);
  f.u[1] = *(const uint4*)(p + gapElems);
  return f.v;
}

// C[M_ x NT_] = Xbf[M_ x 512] * WT^T ; stores bf16 into qkv[3][M_][512], k gets +bk.
__global__ __launch_bounds__(256) void gemm_qkv_bf16(const __bf16* __restrict__ xbf,
                                                     const __bf16* __restrict__ wt,
                                                     const float* __restrict__ bk,
                                                     __bf16* __restrict__ qkv) {
  __shared__ __align__(16) __bf16 smem[2][STAGE_ELEMS];   // ping-pong A|B stages, 40 KiB

  const int blockM = blockIdx.y * TILE_M;
  const int blockN = blockIdx.x * TILE_N;
  const int tid   = threadIdx.x;
  const int lane  = tid & 31;
  const int wid   = tid >> 5;
  const int waveM = (wid & 3) * 32;    // 4 waves along M
  const int waveN = (wid >> 2) * 64;   // 2 waves along N
  const int laneN = lane & 15;
  const int khalf = lane >> 4;

  // per-thread tile-copy assignment: 2 chunks of A + 2 chunks of B, 16B each
  const int row0 = tid >> 2,            cc0 = tid & 3;
  const int row1 = (tid + 256) >> 2,    cc1 = (tid + 256) & 3;
  const __bf16* gA0 = xbf + (size_t)(blockM + row0) * D_ + cc0 * 8;
  const __bf16* gA1 = xbf + (size_t)(blockM + row1) * D_ + cc1 * 8;
  const __bf16* gB0 = wt  + (size_t)(blockN + row0) * D_ + cc0 * 8;
  const __bf16* gB1 = wt  + (size_t)(blockN + row1) * D_ + cc1 * 8;
  const unsigned lA0 = (unsigned)(size_t)&smem[0][0] + (unsigned)(row0 * LDSK + cc0 * 8) * 2u;
  const unsigned lA1 = (unsigned)(size_t)&smem[0][0] + (unsigned)(row1 * LDSK + cc1 * 8) * 2u;
  const unsigned lB0 = lA0 + (unsigned)(TILE_M * LDSK) * 2u;
  const unsigned lB1 = lA1 + (unsigned)(TILE_M * LDSK) * 2u;
  const unsigned stageB = (unsigned)(STAGE_ELEMS * 2);   // byte size of one stage

  v8f acc[2][4];
  {
    v8f z = {0.f, 0.f, 0.f, 0.f, 0.f, 0.f, 0.f, 0.f};
#pragma unroll
    for (int mt = 0; mt < 2; ++mt)
#pragma unroll
      for (int nt = 0; nt < 4; ++nt) acc[mt][nt] = z;
  }

  // prologue: async-fetch k-tile 0 into stage 0
  async_copy16(lA0, gA0);
  async_copy16(lA1, gA1);
  async_copy16(lB0, gB0);
  async_copy16(lB1, gB1);

  for (int step = 0; step < KSTEPS; ++step) {
    wait_async0();        // my async writes for stage (step&1) have landed
    __syncthreads();      // everyone's have landed

    if (step + 1 < KSTEPS) {     // uniform branch: prefetch next tile into other stage
      int ko = (step + 1) * KB;
      unsigned so = ((step + 1) & 1) ? stageB : 0u;
      async_copy16(lA0 + so, gA0 + ko);
      async_copy16(lA1 + so, gA1 + ko);
      async_copy16(lB0 + so, gB0 + ko);
      async_copy16(lB1 + so, gB1 + ko);
    }

    const __bf16* sA = smem[step & 1];
    const __bf16* sB = sA + TILE_M * LDSK;

    v16bf afrag[2], bfrag[4];
#pragma unroll
    for (int mt = 0; mt < 2; ++mt)   // A 16x32: lane<16 -> K{0..7,16..23}, lane>=16 -> K{8..15,24..31}
      afrag[mt] = load_frag(&sA[(waveM + mt * 16 + laneN) * LDSK + khalf * 8], 16);
#pragma unroll
    for (int nt = 0; nt < 4; ++nt)   // B 32x16: lane<16 -> K{0..7,16..23}, lane>=16 -> K{8..15,24..31}
      bfrag[nt] = load_frag(&sB[(waveN + nt * 16 + laneN) * LDSK + khalf * 8], 16);

#pragma unroll
    for (int mt = 0; mt < 2; ++mt)
#pragma unroll
      for (int nt = 0; nt < 4; ++nt)
        acc[mt][nt] = __builtin_amdgcn_wmma_f32_16x16x32_bf16(
            false, afrag[mt], false, bfrag[nt], (short)0, acc[mt][nt], false, false);

    __syncthreads();      // done reading this stage before it is overwritten at step+2
  }

  // ---- epilogue ----
  // A 128-wide N tile never crosses a 512 boundary, so `which` is block-uniform.
  const int which = blockN >> 9;
  const int nbase = (blockN & (D_ - 1)) + waveN;
  __bf16* outb = qkv + (size_t)which * M_ * D_;
  const int rowBase = (lane < 16) ? 0 : 8;   // C/D: lane 0-15 -> M r, lane 16-31 -> M 8+r

#pragma unroll
  for (int nt = 0; nt < 4; ++nt) {
    const int n = nbase + nt * 16 + laneN;
    const float bias = (which == 1) ? bk[n] : 0.0f;   // uniform branch, one load per nt
#pragma unroll
    for (int mt = 0; mt < 2; ++mt) {
#pragma unroll
      for (int r = 0; r < 8; ++r) {
        const int gm = blockM + waveM + mt * 16 + rowBase + r;
        outb[(size_t)gm * D_ + n] = (__bf16)(acc[mt][nt][r] + bias);
      }
    }
  }
}

// ----------------------------------------------------- windowed attention ---

__global__ __launch_bounds__(256) void rel_attn_window(const __bf16* __restrict__ q,
                                                       const __bf16* __restrict__ k,
                                                       const __bf16* __restrict__ v,
                                                       const float* __restrict__ pos,
                                                       float* __restrict__ out) {
  __shared__ float spos[DH_ * W_];
  for (int i = threadIdx.x; i < DH_ * W_; i += blockDim.x) spos[i] = pos[i];
  __syncthreads();

  size_t idx = (size_t)blockIdx.x * blockDim.x + threadIdx.x;  // over M_*D_
  int c  = (int)(idx & (D_ - 1));
  int s  = (int)((idx >> 9) & (S_ - 1));
  int dh = c & (DH_ - 1);

  float kv = (float)k[idx];
  int nv = S_ - s; nv = nv < W_ ? nv : W_;   // number of valid window slots

  float l[W_];
  float m = -INFINITY;
#pragma unroll
  for (int i = 0; i < W_; ++i) {
    if (i < nv) {
      float qv = (float)q[idx + (size_t)i * D_];
      l[i] = (qv + spos[dh * W_ + i]) * kv;
      m = fmaxf(m, l[i]);
    }
  }
  float num = 0.f, den = 0.f;
#pragma unroll
  for (int i = 0; i < W_; ++i) {
    if (i < nv) {
      float e = __expf(l[i] - m);
      den += e;
      num += e * (float)v[idx + (size_t)i * D_];
    }
  }
  out[idx] = num / den;
}

// ------------------------------------------------------------------ launch --

extern "C" void kernel_launch(void* const* d_in, const int* in_sizes, int n_in,
                              void* d_out, int out_size, void* d_ws, size_t ws_size,
                              hipStream_t stream) {
  const float* x   = (const float*)d_in[0];
  const float* Wq  = (const float*)d_in[1];
  const float* Wk  = (const float*)d_in[2];
  const float* bk  = (const float*)d_in[3];
  const float* Wv  = (const float*)d_in[4];
  const float* pos = (const float*)d_in[5];
  float* out = (float*)d_out;

  // workspace carve-up
  char* ws = (char*)d_ws;
  __bf16* xbf = (__bf16*)ws;                                    //  32 MiB
  __bf16* wt  = (__bf16*)(ws + (size_t)M_ * D_ * 2);            //  1.5 MiB
  __bf16* qkv = (__bf16*)(ws + (size_t)M_ * D_ * 2
                             + (size_t)NT_ * D_ * 2);           //  96 MiB
  const __bf16* qb = qkv;
  const __bf16* kb = qkv + (size_t)M_ * D_;
  const __bf16* vb = qkv + (size_t)2 * M_ * D_;

  // 1) fp32 -> bf16 conversions
  convert_x_bf16<<<(M_ * D_ / 4) / 256, 256, 0, stream>>>(x, xbf);
  convert_wt_bf16<<<(NT_ * D_) / 256, 256, 0, stream>>>(Wq, Wk, Wv, wt);

  // 2) fused q|k|v GEMM: (32768 x 512) * (512 x 1536), bf16 WMMA, fp32 acc,
  //    async global->LDS double-buffered pipeline
  dim3 grid(NT_ / TILE_N, M_ / TILE_M, 1);
  gemm_qkv_bf16<<<grid, 256, 0, stream>>>(xbf, wt, bk, qkv);

  // 3) per-channel windowed softmax-attention
  rel_attn_window<<<(M_ * D_) / 256, 256, 0, stream>>>(qb, kb, vb, pos, out);
}
